// LGSBlock_34351148433790
// MI455X (gfx1250) — matrix operand order repaired and checked by reading, not compiled, not run on hardware
//
#include <hip/hip_runtime.h>
#include <hip/hip_bf16.h>
#include <math.h>

#define ROWS   9216        // 4*2304 == 2*4608
#define CD     192
#define DIN    384
#define DST    64
#define NHD    6
#define HDIM   64
#define ZXD    902
#define CHK    128
#define NC     36
#define SEQ    4608

typedef __attribute__((ext_vector_type(16))) _Float16 v16h;
typedef __attribute__((ext_vector_type(8)))  float    v8f;

__device__ __forceinline__ v8f wmma_f(v16h a, v16h b, v8f c) {
  return __builtin_amdgcn_wmma_f32_16x16x32_f16(false, a, false, b, (short)0, c, false, false);
}
__device__ __forceinline__ v8f zero8() {
  v8f z;
#pragma unroll
  for (int i = 0; i < 8; i++) z[i] = 0.f;
  return z;
}
__device__ __forceinline__ v16h zfrag() {
  v16h z;
#pragma unroll
  for (int i = 0; i < 16; i++) z[i] = (_Float16)0.f;
  return z;
}
// A fragment: row-major f32 row pointer already offset to column kb; K run contiguous.
__device__ __forceinline__ v16h afrag(const float* rowk, int half) {
  v16h a;
#pragma unroll
  for (int i = 0; i < 8; i++) {
    a[i]     = (_Float16)rowk[half * 16 + i];
    a[8 + i] = (_Float16)rowk[half * 16 + 8 + i];
  }
  return a;
}
// B fragment: p points at Bmat[kb + half*16][col]; stride between consecutive k in floats.
__device__ __forceinline__ v16h bfrag(const float* p, int stride) {
  v16h b;
#pragma unroll
  for (int i = 0; i < 16; i++) b[i] = (_Float16)p[(size_t)i * stride];
  return b;
}

// ---------------- generic GEMM: out = act(A[M,K] @ W[N,K]^T + bias) + resid ----------------
// act: 0 none, 1 gelu(exact), 2 silu, 3 sigmoid
// GUARD=false fast path: N % 32 == 0, two N tiles per wave sharing one A fragment.
template <bool GUARD>
__global__ void gemm_wmma_kernel(const float* __restrict__ A, const float* __restrict__ W,
                                 const float* __restrict__ bias, const float* __restrict__ resid,
                                 float* __restrict__ out, int M, int N, int K, int act) {
  const int lane = threadIdx.x & 31;
  const int wave = threadIdx.x >> 5;
  const int half = lane >> 4, l15 = lane & 15;
  const int tM = blockIdx.y * 4 + wave;
  if (tM * 16 >= M) return;
  const int mA = tM * 16 + l15;

  if (!GUARD) {
    const int n0 = blockIdx.x * 32 + l15;   // tile 0 col
    const int n1 = n0 + 16;                 // tile 1 col
    v8f acc0 = zero8(), acc1 = zero8();
    const float* Arow = A + (size_t)mA * K;
    const float* W0 = W + (size_t)n0 * K + half * 16;
    const float* W1 = W + (size_t)n1 * K + half * 16;
    for (int kb = 0; kb < K; kb += 32) {
      v16h a  = afrag(Arow + kb, half);
      v16h b0 = bfrag(W0 + kb, 1);
      v16h b1 = bfrag(W1 + kb, 1);
      acc0 = wmma_f(a, b0, acc0);
      acc1 = wmma_f(a, b1, acc1);
    }
#pragma unroll
    for (int t = 0; t < 2; t++) {
      v8f acc = t ? acc1 : acc0;
      const int nn = blockIdx.x * 32 + t * 16 + l15;
      float bi = bias ? bias[nn] : 0.f;
#pragma unroll
      for (int r = 0; r < 8; r++) {
        const int mm = tM * 16 + r + 8 * half;
        float v = acc[r] + bi;
        if (act == 1)      v = 0.5f * v * (1.f + erff(v * 0.70710678f));
        else if (act == 2) v = v / (1.f + __expf(-v));
        else if (act == 3) v = 1.f / (1.f + __expf(-v));
        if (resid) v += resid[(size_t)mm * N + nn];
        out[(size_t)mm * N + nn] = v;
      }
    }
    return;
  }

  // guarded single-tile path (N = 902 or 6)
  const int tN = blockIdx.x;
  const int nB = tN * 16 + l15;
  const bool nv = nB < N;
  v8f acc = zero8();
  for (int kb = 0; kb < K; kb += 32) {
    v16h a = afrag(A + (size_t)mA * K + kb, half);
    v16h b = nv ? bfrag(W + (size_t)nB * K + kb + half * 16, 1) : zfrag();
    acc = wmma_f(a, b, acc);
  }
#pragma unroll
  for (int r = 0; r < 8; r++) {
    int mm = tM * 16 + r + 8 * half;
    int nn = tN * 16 + l15;
    if (nn < N) {
      float v = acc[r];
      if (bias) v += bias[nn];
      if (act == 1)      v = 0.5f * v * (1.f + erff(v * 0.70710678f));
      else if (act == 2) v = v / (1.f + __expf(-v));
      else if (act == 3) v = 1.f / (1.f + __expf(-v));
      if (resid) v += resid[(size_t)mm * N + nn];
      out[(size_t)mm * N + nn] = v;
    }
  }
}

// ---------------- LayerNorm (C=192), optional interleaved xs output ----------------
__global__ void layernorm_kernel(const float* __restrict__ x, const float* __restrict__ w,
                                 const float* __restrict__ b, float* __restrict__ out,
                                 float* __restrict__ xs_out) {
  const int row = blockIdx.x;   // b4*2304 + l
  const int tid = threadIdx.x;  // 64 threads
  __shared__ float rs[64], rq[64];
  const float v0 = x[(size_t)row * CD + tid];
  const float v1 = x[(size_t)row * CD + tid + 64];
  const float v2 = x[(size_t)row * CD + tid + 128];
  rs[tid] = v0 + v1 + v2;
  rq[tid] = v0 * v0 + v1 * v1 + v2 * v2;
  __syncthreads();
  for (int s = 32; s > 0; s >>= 1) {
    if (tid < s) { rs[tid] += rs[tid + s]; rq[tid] += rq[tid + s]; }
    __syncthreads();
  }
  const float mean = rs[0] / (float)CD;
  const float var  = rq[0] / (float)CD - mean * mean;
  const float inv  = rsqrtf(var + 1e-5f);
  const int b4 = row / 2304, l = row % 2304;
  const size_t xsrow = (size_t)(b4 & 1) * SEQ + 2 * l + (b4 >> 1);
  const float vv[3] = {v0, v1, v2};
#pragma unroll
  for (int i = 0; i < 3; i++) {
    const int c = tid + i * 64;
    const float o = (vv[i] - mean) * inv * w[c] + b[c];
    out[(size_t)row * CD + c] = o;
    if (xs_out) xs_out[xsrow * CD + c] = o;
  }
}

// ---------------- conv4 + SiLU split + dt/a ----------------
__global__ void conv_silu_kernel(const float* __restrict__ zx, const float* __restrict__ cw,
                                 const float* __restrict__ cb, const float* __restrict__ dtb,
                                 const float* __restrict__ Alog, float* __restrict__ xp,
                                 float* __restrict__ Xsc, float* __restrict__ Bm,
                                 float* __restrict__ Cm, float* __restrict__ a_buf) {
  const int row = blockIdx.x;       // bh*4608 + t
  const int bh = row / SEQ, t = row % SEQ;
  const int ch = threadIdx.x;       // 512
  __shared__ float dt_sh[NHD];
  if (ch < NHD) {
    float xr = zx[(size_t)row * ZXD + 896 + ch] + dtb[ch];
    float dt = (xr > 20.f) ? xr : log1pf(__expf(xr));
    dt_sh[ch] = dt;
    a_buf[(size_t)row * NHD + ch] = -__expf(Alog[ch]) * dt;
  }
  __syncthreads();
  float acc = cb[ch];
#pragma unroll
  for (int j = 0; j < 4; j++) {
    int tt = t - 3 + j;
    if (tt >= 0) acc += cw[ch * 4 + j] * zx[((size_t)bh * SEQ + tt) * ZXD + 384 + ch];
  }
  float v = acc / (1.f + __expf(-acc));  // silu
  if (ch < DIN) {
    xp[(size_t)row * DIN + ch] = v;
    Xsc[(size_t)row * DIN + ch] = v * dt_sh[ch >> 6];
  } else if (ch < DIN + DST) {
    Bm[(size_t)row * DST + ch - DIN] = v;
  } else {
    Cm[(size_t)row * DST + ch - DIN - DST] = v;
  }
}

// ---------------- SSD pass 1: per-chunk states = (decay*X)^T @ B  (64x64, K=128) ----------------
__global__ void ssd_states_kernel(const float* __restrict__ Xsc, const float* __restrict__ Bm,
                                  const float* __restrict__ a_buf, float* __restrict__ states,
                                  float* __restrict__ csum) {
  __shared__ float acs[CHK];
  __shared__ float dec[CHK];
  const int c = blockIdx.x, h = blockIdx.y, bh = blockIdx.z;
  const int t0 = c * CHK;
  const int tid = threadIdx.x;  // 256
  if (tid < CHK) acs[tid] = a_buf[((size_t)bh * SEQ + t0 + tid) * NHD + h];
  __syncthreads();
  if (tid == 0) { float s = 0.f; for (int i = 0; i < CHK; i++) { s += acs[i]; acs[i] = s; } }
  __syncthreads();
  const float Ac = acs[CHK - 1];
  if (tid < CHK) dec[tid] = __expf(Ac - acs[tid]);
  if (tid == 0) csum[(bh * NHD + h) * NC + c] = Ac;
  __syncthreads();
  const int lane = tid & 31, wave = tid >> 5, half = lane >> 4, l15 = lane & 15;
  const float* Xb = Xsc + (((size_t)bh * SEQ + t0) * NHD + h) * HDIM;  // [l][p] stride DIN
  const float* Bb = Bm + ((size_t)bh * SEQ + t0) * DST;                // [l][n] stride DST
  float* S = states + ((size_t)(bh * NHD + h) * NC + c) * 64 * 64;
  for (int tIdx = wave; tIdx < 16; tIdx += 8) {
    const int pt = tIdx >> 2, nt = tIdx & 3;
    const int p = pt * 16 + l15, n = nt * 16 + l15;
    v8f acc = zero8();
    for (int kb = 0; kb < CHK; kb += 32) {
      v16h a, b;
#pragma unroll
      for (int i = 0; i < 8; i++) {
        int k0 = kb + half * 16 + i;
        int k1 = k0 + 8;
        a[i]     = (_Float16)(dec[k0] * Xb[(size_t)k0 * DIN + p]);
        a[8 + i] = (_Float16)(dec[k1] * Xb[(size_t)k1 * DIN + p]);
      }
#pragma unroll
      for (int i = 0; i < 16; i++) {
        int k = kb + half * 16 + i;
        b[i] = (_Float16)Bb[(size_t)k * DST + n];
      }
      acc = wmma_f(a, b, acc);
    }
#pragma unroll
    for (int r = 0; r < 8; r++)
      S[(size_t)(pt * 16 + r + 8 * half) * 64 + nt * 16 + l15] = acc[r];
  }
}

// ---------------- SSD pass 2: sequential inter-chunk scan ----------------
__global__ void ssd_scan_kernel(const float* __restrict__ states, const float* __restrict__ csum,
                                float* __restrict__ prev) {
  const int bhh = blockIdx.x;   // bh*6+h
  const int tid = threadIdx.x;  // 256, 16 elems each
  float P[16];
#pragma unroll
  for (int i = 0; i < 16; i++) P[i] = 0.f;
  for (int c = 0; c < NC; c++) {
    const float* S = states + ((size_t)bhh * NC + c) * 4096;
    float* Pr = prev + ((size_t)bhh * NC + c) * 4096;
    const float e = __expf(csum[bhh * NC + c]);
#pragma unroll
    for (int i = 0; i < 16; i++) {
      const int idx = tid * 16 + i;
      Pr[idx] = P[i];
      P[i] = e * P[i] + S[idx];
    }
  }
}

// ---------------- SSD pass 3: Y = (L o C@B^T) @ X + e^acs * C @ prev^T + D*xp ----------------
__global__ void ssd_y_kernel(const float* __restrict__ Xsc, const float* __restrict__ Bm,
                             const float* __restrict__ Cm, const float* __restrict__ a_buf,
                             const float* __restrict__ prev, const float* __restrict__ xp,
                             const float* __restrict__ Dw, float* __restrict__ y) {
  __shared__ _Float16 Pst[8][16][CHK];
  __shared__ float acs[CHK];
  const int c = blockIdx.x, h = blockIdx.y, bh = blockIdx.z;
  const int t0 = c * CHK;
  const int tid = threadIdx.x;  // 256
  if (tid < CHK) acs[tid] = a_buf[((size_t)bh * SEQ + t0 + tid) * NHD + h];
  __syncthreads();
  if (tid == 0) { float s = 0.f; for (int i = 0; i < CHK; i++) { s += acs[i]; acs[i] = s; } }
  __syncthreads();
  const int lane = tid & 31, wave = tid >> 5, half = lane >> 4, l15 = lane & 15;
  const float* Cb = Cm + ((size_t)bh * SEQ + t0) * DST;
  const float* Bb = Bm + ((size_t)bh * SEQ + t0) * DST;
  const float* Xb = Xsc + (((size_t)bh * SEQ + t0) * NHD + h) * HDIM;
  const float* Pv = prev + ((size_t)(bh * NHD + h) * NC + c) * 64 * 64;
  const int lA = wave * 16 + l15;
  // G strip: rows [wave*16, +16), all 128 cols; masked+decayed -> f16 P strip in LDS
  for (int st = 0; st < 8; st++) {
    v8f g = zero8();
    for (int kb = 0; kb < DST; kb += 32) {
      v16h a = afrag(Cb + (size_t)lA * DST + kb, half);
      v16h b = bfrag(Bb + (size_t)(st * 16 + l15) * DST + kb + half * 16, 1);
      g = wmma_f(a, b, g);
    }
#pragma unroll
    for (int r = 0; r < 8; r++) {
      int l = wave * 16 + r + 8 * half;
      int s = st * 16 + l15;
      float v = (s <= l) ? g[r] * __expf(acs[l] - acs[s]) : 0.f;
      Pst[wave][r + 8 * half][s] = (_Float16)v;
    }
  }
  __syncthreads();
  const float Dh = Dw[h];
  for (int pt = 0; pt < 4; pt++) {
    v8f acc = zero8();
    // Y_diag = P @ X, K=128
    for (int kb = 0; kb < CHK; kb += 32) {
      v16h a;
      const _Float16* ap = &Pst[wave][l15][kb];
#pragma unroll
      for (int i = 0; i < 8; i++) { a[i] = ap[half * 16 + i]; a[8 + i] = ap[half * 16 + 8 + i]; }
      v16h b;
      const int p = pt * 16 + l15;
#pragma unroll
      for (int i = 0; i < 16; i++) b[i] = (_Float16)Xb[(size_t)(kb + half * 16 + i) * DIN + p];
      acc = wmma_f(a, b, acc);
    }
    // Y_off = (e^acs * C) @ prev^T, K=64
    const float el = __expf(acs[lA]);
    for (int kb = 0; kb < DST; kb += 32) {
      v16h a;
      const float* cp = Cb + (size_t)lA * DST + kb;
#pragma unroll
      for (int i = 0; i < 8; i++) {
        a[i]     = (_Float16)(el * cp[half * 16 + i]);
        a[8 + i] = (_Float16)(el * cp[half * 16 + 8 + i]);
      }
      v16h b = bfrag(Pv + (size_t)(pt * 16 + l15) * 64 + kb + half * 16, 1);
      acc = wmma_f(a, b, acc);
    }
#pragma unroll
    for (int r = 0; r < 8; r++) {
      int l = wave * 16 + r + 8 * half;
      int p = pt * 16 + l15;
      size_t gl = (size_t)bh * SEQ + t0 + l;
      y[(gl * NHD + h) * HDIM + p] = acc[r] + Dh * xp[gl * DIN + h * HDIM + p];
    }
  }
}

// ---------------- mamba gate (silu(z)) + RMSNorm ----------------
__global__ void mamba_gate_norm_kernel(const float* __restrict__ y, const float* __restrict__ zx,
                                       const float* __restrict__ nw, float* __restrict__ out) {
  const int row = blockIdx.x;   // bh*4608 + l
  const int tid = threadIdx.x;  // 128, 3 elems each
  __shared__ float red[128];
  float vals[3];
  float acc = 0.f;
#pragma unroll
  for (int i = 0; i < 3; i++) {
    const int cc = tid + i * 128;
    const float yv = y[(size_t)row * DIN + cc];
    const float zv = zx[(size_t)row * ZXD + cc];
    const float g = yv * (zv / (1.f + __expf(-zv)));
    vals[i] = g;
    acc += g * g;
  }
  red[tid] = acc;
  __syncthreads();
  for (int s = 64; s > 0; s >>= 1) {
    if (tid < s) red[tid] += red[tid + s];
    __syncthreads();
  }
  const float r = rsqrtf(red[0] / (float)DIN + 1e-5f);
#pragma unroll
  for (int i = 0; i < 3; i++) {
    const int cc = tid + i * 128;
    out[(size_t)row * DIN + cc] = vals[i] * r * nw[cc];
  }
}

// ---------------- window partition with roll(-4,-4) ----------------
__global__ void window_gather_kernel(const float* __restrict__ xn, float* __restrict__ xw) {
  size_t i = (size_t)blockIdx.x * blockDim.x + threadIdx.x;
  if (i >= (size_t)ROWS * CD) return;
  const int c = i % CD;
  const int row = i / CD;
  const int b = row / 2304, r = row % 2304;
  const int wi = r / 64, t = r % 64;
  const int wrow = wi / 6, wcol = wi % 6;
  const int ii = t >> 3, jj = t & 7;
  const int hh = (wrow * 8 + ii + 4) % 48;
  const int ww = (wcol * 8 + jj + 4) % 48;
  xw[i] = xn[((size_t)b * 2304 + hh * 48 + ww) * CD + c];
}

// ---------------- fused window attention: WMMA QK^T + mask + softmax + WMMA PV + gate ----------------
__global__ void attn_kernel(const float* __restrict__ qkv, const float* __restrict__ gate,
                            float* __restrict__ ctx) {
  __shared__ _Float16 Pst[4][16][64];
  const int blk = blockIdx.x;  // 144*6
  const int win = blk / NHD, head = blk % NHD;
  const int wi = win % 36, wrow = wi / 6, wcol = wi % 6;
  const int tid = threadIdx.x, lane = tid & 31, wave = tid >> 5;
  const int half = lane >> 4, l15 = lane & 15;
  const float* base = qkv + (size_t)win * 64 * 576;
  const int m = wave * 16 + l15;
  v16h qa;
  {
    const float* qp = base + (size_t)m * 576 + head * 32;
#pragma unroll
    for (int i = 0; i < 8; i++) {
      qa[i]     = (_Float16)(0.17677670f * qp[half * 16 + i]);
      qa[8 + i] = (_Float16)(0.17677670f * qp[half * 16 + 8 + i]);
    }
  }
  v8f s[4];
  for (int st = 0; st < 4; st++) {
    const float* kp = base + (size_t)(st * 16 + l15) * 576 + 192 + head * 32 + half * 16;
    v16h b;
#pragma unroll
    for (int i = 0; i < 16; i++) b[i] = (_Float16)kp[i];
    s[st] = wmma_f(qa, b, zero8());
  }
  float rowsum[8];
#pragma unroll
  for (int r = 0; r < 8; r++) {
    const int t1 = wave * 16 + r + 8 * half;
    const int h1 = wrow * 8 + (t1 >> 3), w1 = wcol * 8 + (t1 & 7);
    const int rh1 = (h1 < 40) ? 0 : ((h1 < 44) ? 1 : 2);
    const int rw1 = (w1 < 40) ? 0 : ((w1 < 44) ? 1 : 2);
    float sv[4], mx = -1e30f;
    for (int st = 0; st < 4; st++) {
      const int t2 = st * 16 + l15;
      const int h2 = wrow * 8 + (t2 >> 3), w2 = wcol * 8 + (t2 & 7);
      const int rh2 = (h2 < 40) ? 0 : ((h2 < 44) ? 1 : 2);
      const int rw2 = (w2 < 40) ? 0 : ((w2 < 44) ? 1 : 2);
      float v = s[st][r] + ((rh1 == rh2 && rw1 == rw2) ? 0.f : -100.f);
      sv[st] = v;
      mx = fmaxf(mx, v);
    }
    for (int d = 1; d < 16; d <<= 1) mx = fmaxf(mx, __shfl_xor(mx, d, 32));
    float sum = 0.f;
    for (int st = 0; st < 4; st++) {
      float e = __expf(sv[st] - mx);
      sum += e;
      Pst[wave][r + 8 * half][st * 16 + l15] = (_Float16)e;
    }
    for (int d = 1; d < 16; d <<= 1) sum += __shfl_xor(sum, d, 32);
    rowsum[r] = sum;
  }
  __syncthreads();
  for (int pt = 0; pt < 2; pt++) {
    v8f acc = zero8();
    for (int kb = 0; kb < 64; kb += 32) {
      v16h a;
      const _Float16* ap = &Pst[wave][l15][kb];
#pragma unroll
      for (int i = 0; i < 8; i++) { a[i] = ap[half * 16 + i]; a[8 + i] = ap[half * 16 + 8 + i]; }
      v16h b;
      const int d = pt * 16 + l15;
      const float* vp = base + (size_t)(kb + half * 16) * 576 + 384 + head * 32 + d;
#pragma unroll
      for (int i = 0; i < 16; i++) b[i] = (_Float16)vp[(size_t)i * 576];
      acc = wmma_f(a, b, acc);
    }
#pragma unroll
    for (int r = 0; r < 8; r++) {
      const int t1 = wave * 16 + r + 8 * half;
      const int d = pt * 16 + l15;
      const float g = gate[(size_t)(win * 64 + t1) * NHD + head];
      ctx[((size_t)win * 64 + t1) * CD + head * 32 + d] = acc[r] / rowsum[r] * g;
    }
  }
}

// ---------------- fusion input build: [un-interleaved mamba | un-windowed attn] ----------------
__global__ void fusion_prep_kernel(const float* __restrict__ ym, const float* __restrict__ projo,
                                   float* __restrict__ xf_in) {
  size_t i = (size_t)blockIdx.x * blockDim.x + threadIdx.x;
  if (i >= (size_t)ROWS * 2 * CD) return;
  const int c = i % (2 * CD);
  const int row = i / (2 * CD);
  const int b4 = row / 2304, l = row % 2304;
  float v;
  if (c < CD) {
    const int bh = (b4 < 2) ? b4 : b4 - 2;
    const int s = (b4 < 2) ? 0 : 1;
    v = ym[((size_t)bh * SEQ + 2 * l + s) * CD + c];
  } else {
    const int hh = l / 48, ww = l % 48;
    const int h2 = (hh + 44) % 48, w2 = (ww + 44) % 48;
    const int wrow = h2 >> 3, ii = h2 & 7, wcol = w2 >> 3, jj = w2 & 7;
    const size_t arow = (((size_t)b4 * 6 + wrow) * 6 + wcol) * 64 + ii * 8 + jj;
    v = projo[arow * CD + (c - CD)];
  }
  xf_in[i] = v;
}

// ---------------- ECA: pool over L, 3-tap sigmoid channel attention ----------------
__global__ void eca_pool_kernel(const float* __restrict__ xf, const float* __restrict__ ew,
                                float* __restrict__ ca) {
  const int b = blockIdx.x;
  const int c = threadIdx.x;  // 192
  __shared__ float pool[CD];
  float s = 0.f;
  for (int l = 0; l < 2304; l++) s += xf[((size_t)b * 2304 + l) * CD + c];
  pool[c] = s / 2304.f;
  __syncthreads();
  const float pm = (c > 0) ? pool[c - 1] : 0.f;
  const float pp = (c < CD - 1) ? pool[c + 1] : 0.f;
  const float v = ew[0] * pm + ew[1] * pool[c] + ew[2] * pp;
  ca[b * CD + c] = 1.f / (1.f + __expf(-v));
}

__global__ void residual_eca_kernel(const float* __restrict__ x, const float* __restrict__ xf,
                                    const float* __restrict__ ca, float* __restrict__ xo) {
  size_t i = (size_t)blockIdx.x * blockDim.x + threadIdx.x;
  if (i >= (size_t)ROWS * CD) return;
  const int c = i % CD;
  const int b = (int)(i / CD) / 2304;
  xo[i] = x[i] + xf[i] * ca[b * CD + c];
}

// =====================================================================================
extern "C" void kernel_launch(void* const* d_in, const int* in_sizes, int n_in,
                              void* d_out, int out_size, void* d_ws, size_t ws_size,
                              hipStream_t stream) {
  (void)in_sizes; (void)n_in; (void)out_size; (void)ws_size;
  const float* x      = (const float*)d_in[0];
  const float* n1_w   = (const float*)d_in[3];
  const float* n1_b   = (const float*)d_in[4];
  const float* m_in_w = (const float*)d_in[5];
  const float* m_cw   = (const float*)d_in[6];
  const float* m_cb   = (const float*)d_in[7];
  const float* m_dtb  = (const float*)d_in[8];
  const float* m_Alog = (const float*)d_in[9];
  const float* m_D    = (const float*)d_in[10];
  const float* m_nw   = (const float*)d_in[11];
  const float* m_ow   = (const float*)d_in[12];
  const float* qkv_w  = (const float*)d_in[13];
  const float* qkv_b  = (const float*)d_in[14];
  const float* proj_w = (const float*)d_in[15];
  const float* proj_b = (const float*)d_in[16];
  const float* gate_w = (const float*)d_in[17];
  const float* gate_b = (const float*)d_in[18];
  const float* fus_w  = (const float*)d_in[19];
  const float* fus_b  = (const float*)d_in[20];
  const float* eca_w  = (const float*)d_in[21];
  const float* n2_w   = (const float*)d_in[22];
  const float* n2_b   = (const float*)d_in[23];
  const float* fc1_w  = (const float*)d_in[24];
  const float* fc1_b  = (const float*)d_in[25];
  const float* fc2_w  = (const float*)d_in[26];
  const float* fc2_b  = (const float*)d_in[27];
  float* out = (float*)d_out;

  float* ws = (float*)d_ws;
  size_t off = 0;
  auto alloc = [&](size_t n) { float* p = ws + off; off += n; return p; };
  float* xn     = alloc((size_t)ROWS * CD);
  float* xs     = alloc((size_t)ROWS * CD);
  float* zx     = alloc((size_t)ROWS * ZXD);
  float* xp     = alloc((size_t)ROWS * DIN);
  float* Xsc    = alloc((size_t)ROWS * DIN);
  float* Bmb    = alloc((size_t)ROWS * DST);
  float* Cmb    = alloc((size_t)ROWS * DST);
  float* a_buf  = alloc((size_t)ROWS * NHD);
  float* states = alloc((size_t)2 * NHD * NC * 64 * 64);
  float* csum   = alloc((size_t)2 * NHD * NC);
  float* prev   = alloc((size_t)2 * NHD * NC * 64 * 64);
  float* ybuf   = alloc((size_t)ROWS * DIN);
  float* nrm    = alloc((size_t)ROWS * DIN);
  float* ymb    = alloc((size_t)ROWS * CD);
  float* xw     = alloc((size_t)ROWS * CD);
  float* qkvb   = alloc((size_t)ROWS * 576);
  float* gateb  = alloc((size_t)ROWS * NHD);
  float* ctx    = alloc((size_t)ROWS * CD);
  float* projo  = alloc((size_t)ROWS * CD);
  float* xf_in  = alloc((size_t)ROWS * 2 * CD);
  float* xf     = alloc((size_t)ROWS * CD);
  float* ca     = alloc((size_t)4 * CD);
  float* xo     = alloc((size_t)ROWS * CD);
  float* hin    = alloc((size_t)ROWS * CD);
  float* mlp    = alloc((size_t)ROWS * 768);

  // fast path: N % 32 == 0
  auto gemm = [&](const float* A, const float* W, const float* bias, const float* resid,
                  float* o, int M, int N, int K, int act) {
    dim3 g(N / 32, (M / 16 + 3) / 4);
    gemm_wmma_kernel<false><<<g, 128, 0, stream>>>(A, W, bias, resid, o, M, N, K, act);
  };
  // guarded path: arbitrary N
  auto gemm_g = [&](const float* A, const float* W, const float* bias, const float* resid,
                    float* o, int M, int N, int K, int act) {
    dim3 g((N + 15) / 16, (M / 16 + 3) / 4);
    gemm_wmma_kernel<true><<<g, 128, 0, stream>>>(A, W, bias, resid, o, M, N, K, act);
  };
  auto ew_blocks = [](size_t n) { return (unsigned)((n + 255) / 256); };

  // 1) LN1 -> xn, interleaved xs
  layernorm_kernel<<<ROWS, 64, 0, stream>>>(x, n1_w, n1_b, xn, xs);
  // 2) mamba in_proj (N=902, guarded)
  gemm_g(xs, m_in_w, nullptr, nullptr, zx, ROWS, ZXD, CD, 0);
  // 3) conv + silu + dt
  conv_silu_kernel<<<ROWS, 512, 0, stream>>>(zx, m_cw, m_cb, m_dtb, m_Alog, xp, Xsc, Bmb, Cmb, a_buf);
  // 4-6) SSD
  ssd_states_kernel<<<dim3(NC, NHD, 2), 256, 0, stream>>>(Xsc, Bmb, a_buf, states, csum);
  ssd_scan_kernel<<<2 * NHD, 256, 0, stream>>>(states, csum, prev);
  ssd_y_kernel<<<dim3(NC, NHD, 2), 256, 0, stream>>>(Xsc, Bmb, Cmb, a_buf, prev, xp, m_D, ybuf);
  // 7) gate + RMSNorm
  mamba_gate_norm_kernel<<<ROWS, 128, 0, stream>>>(ybuf, zx, m_nw, nrm);
  // 8) mamba out_proj
  gemm(nrm, m_ow, nullptr, nullptr, ymb, ROWS, CD, DIN, 0);
  // 9) window gather (roll + partition)
  window_gather_kernel<<<ew_blocks((size_t)ROWS * CD), 256, 0, stream>>>(xn, xw);
  // 10) qkv, 11) gate (sigmoid, N=6 guarded)
  gemm(xw, qkv_w, qkv_b, nullptr, qkvb, ROWS, 576, CD, 0);
  gemm_g(xw, gate_w, gate_b, nullptr, gateb, ROWS, NHD, CD, 3);
  // 12) fused attention
  attn_kernel<<<144 * NHD, 128, 0, stream>>>(qkvb, gateb, ctx);
  // 13) attn proj
  gemm(ctx, proj_w, proj_b, nullptr, projo, ROWS, CD, CD, 0);
  // 14) fusion input
  fusion_prep_kernel<<<ew_blocks((size_t)ROWS * 2 * CD), 256, 0, stream>>>(ymb, projo, xf_in);
  // 15) fusion GEMM
  gemm(xf_in, fus_w, fus_b, nullptr, xf, ROWS, CD, 2 * CD, 0);
  // 16) ECA + residual
  eca_pool_kernel<<<4, CD, 0, stream>>>(xf, eca_w, ca);
  residual_eca_kernel<<<ew_blocks((size_t)ROWS * CD), 256, 0, stream>>>(x, xf, ca, xo);
  // 17) LN2
  layernorm_kernel<<<ROWS, 64, 0, stream>>>(xo, n2_w, n2_b, hin, nullptr);
  // 18) MLP: fc1 (GELU) then fc2 (+bias, +residual xo) -> out
  gemm(hin, fc1_w, fc1_b, nullptr, mlp, ROWS, 768, CD, 1);
  gemm(mlp, fc2_w, fc2_b, xo, out, ROWS, CD, 768, 0);
}
